// bicourageInvNonLinear_45105746543038
// MI455X (gfx1250) — matrix operand order repaired and checked by reading, not compiled, not run on hardware
//
#include <hip/hip_runtime.h>
#include <hip/hip_bf16.h>

// ---------------------------------------------------------------------------
// GraphSAGE 2-tower GNN + pooling + MLP for MI455X (gfx1250, wave32, WMMA)
// ---------------------------------------------------------------------------

typedef __attribute__((ext_vector_type(16))) __bf16 v16bf;
typedef __attribute__((ext_vector_type(8)))  float  v8f;

#define GNN_N_NODES 50000
#define GNN_N_EDGES 400000
#define GNN_D_IN    128
#define GNN_GRAPHS  64
#define GNN_POOLDIM 1600
#define GEMM_TM     128     // block row tile (8 waves x 16 rows)

// ---------------------------------------------------------------------------
// Utility: fill a float buffer with a constant (used for zeroing agg/deg).
// ---------------------------------------------------------------------------
__global__ void fill_f32_kernel(float* __restrict__ p, float v, long long n) {
    long long i = (long long)blockIdx.x * blockDim.x + threadIdx.x;
    if (i < n) p[i] = v;
}

// ---------------------------------------------------------------------------
// Degree: deg[c] += 1 for each edge endpoint c, then inv = 1/max(deg,1).
// ---------------------------------------------------------------------------
__global__ void deg_count_kernel(const int* __restrict__ col, float* __restrict__ deg, int E) {
    int e = blockIdx.x * blockDim.x + threadIdx.x;
    if (e < E) atomicAdd(&deg[col[e]], 1.0f);
}

__global__ void deg_inv_kernel(const float* __restrict__ deg, float* __restrict__ inv, int N) {
    int i = blockIdx.x * blockDim.x + threadIdx.x;
    if (i < N) inv[i] = 1.0f / fmaxf(deg[i], 1.0f);
}

// ---------------------------------------------------------------------------
// Edge aggregation: agg[col[e], j] += h[row[e], j].  One thread per (e, j-pair)
// using float2 (all layer widths are even, rows stay 8B aligned).  Gather is
// coalesced within an edge row; scatter uses global_atomic_add_f32.
// ---------------------------------------------------------------------------
__global__ void aggregate_kernel(const float* __restrict__ h,
                                 const int* __restrict__ row,
                                 const int* __restrict__ col,
                                 float* __restrict__ agg,
                                 int d2, int total2) {
    int i = blockIdx.x * blockDim.x + threadIdx.x;
    if (i >= total2) return;
    int e = i / d2;
    int j = i - e * d2;
    int d = d2 << 1;
    float2 v = *(const float2*)(h + (size_t)row[e] * d + (j << 1));
    float* dst = agg + (size_t)col[e] * d + (j << 1);
    atomicAdd(dst, v.x);
    atomicAdd(dst + 1, v.y);
}

// ---------------------------------------------------------------------------
// LDS tile staging for the WMMA GEMM (f32 -> bf16 conversion fused).
// Fast path: full tile, branch-free float2 loads; slow path: guarded.
// HAS_SCALE resolved at compile time (neighbor half multiplies rows by deg^-1).
// Row pitch 34 (= 17 dwords, odd) -> conflict-free fragment gathers.
// ---------------------------------------------------------------------------
template<bool HAS_SCALE>
__device__ __forceinline__
void stage_tiles(const float* __restrict__ A, const float* __restrict__ rowScale,
                 const float* __restrict__ W,
                 __bf16 (* __restrict__ sA)[34], __bf16 (* __restrict__ sB)[34],
                 int rowBase, int colBase, int kb, int N, int K, int M, int tid) {
    // ---- A tile: GEMM_TM rows x 32 k ----
    if (rowBase + GEMM_TM <= N && kb + 32 <= K) {
        const float* Ab = A + (size_t)rowBase * K + kb;
#pragma unroll
        for (int p = tid; p < GEMM_TM * 16; p += 256) {
            int r  = p >> 4;
            int kp = (p & 15) << 1;
            float2 v = *(const float2*)(Ab + (size_t)r * K + kp);
            float sc = HAS_SCALE ? rowScale[rowBase + r] : 1.0f;
            sA[r][kp]     = (__bf16)(v.x * sc);
            sA[r][kp + 1] = (__bf16)(v.y * sc);
        }
    } else {
        for (int p = tid; p < GEMM_TM * 16; p += 256) {
            int r  = p >> 4;
            int kp = (p & 15) << 1;
            int gr = rowBase + r;
            float v0 = 0.0f, v1 = 0.0f;
            if (gr < N) {
                float sc = HAS_SCALE ? rowScale[gr] : 1.0f;
                int k0 = kb + kp;
                if (k0     < K) v0 = A[(size_t)gr * K + k0] * sc;
                if (k0 + 1 < K) v1 = A[(size_t)gr * K + k0 + 1] * sc;
            }
            sA[r][kp]     = (__bf16)v0;
            sA[r][kp + 1] = (__bf16)v1;
        }
    }
    // ---- W tile: 32 k x 32 cols (scalar loads: M may be odd) ----
    if (kb + 32 <= K && colBase + 32 <= M) {
#pragma unroll
        for (int p = tid; p < 32 * 16; p += 256) {
            int r  = p >> 4;
            int cp = (p & 15) << 1;
            const float* Wr = W + (size_t)(kb + r) * M + colBase + cp;
            sB[r][cp]     = (__bf16)Wr[0];
            sB[r][cp + 1] = (__bf16)Wr[1];
        }
    } else {
        for (int p = tid; p < 32 * 16; p += 256) {
            int r  = p >> 4;
            int cp = (p & 15) << 1;
            int gk = kb + r;
            int gc = colBase + cp;
            float v0 = 0.0f, v1 = 0.0f;
            if (gk < K) {
                if (gc     < M) v0 = W[(size_t)gk * M + gc];
                if (gc + 1 < M) v1 = W[(size_t)gk * M + gc + 1];
            }
            sB[r][cp]     = (__bf16)v0;
            sB[r][cp + 1] = (__bf16)v1;
        }
    }
}

// ---------------------------------------------------------------------------
// Fused WMMA GEMM:  OUT[r, c] = relu( (A[r,:] * rowScale[r]) @ W[:, c] + bias[c] )
//
// Block = 256 threads = 8 waves; block tile = 128 rows x 32 cols.
// Each wave owns 16 rows x 32 cols = two 16x16 D tiles sharing one A fragment
// -> 2x v_wmma_f32_16x16x32_bf16 per K-chunk, one barrier per chunk via
// double-buffered LDS (stage chunk i+1 while computing chunk i).
//
// Fragment layouts per CDNA5 ISA 7.12.2 (wave32):
//   A 16x32 bf16: lane m=lane&15, half=lane>>4; vgpr v packs K = k0,k0+1 with
//                 k0 = (v&3)*2 + (v>>2)*16 + half*8
//   B 32x16 bf16: lane n = lane&15, half=lane>>4; vgpr v packs K = k0,k0+1 (k0 as for A)
//   C/D 16x16 f32: vgpr v -> M = v + 8*half, N = lane&15
// ---------------------------------------------------------------------------
template<bool HAS_SCALE>
__global__ __launch_bounds__(256)
void sage_gemm_bias_relu_kernel(const float* __restrict__ A,
                                const float* __restrict__ rowScale,
                                const float* __restrict__ W,
                                const float* __restrict__ bias,
                                float* __restrict__ OUT,
                                int N, int K, int M, int ldc) {
    __shared__ __bf16 sA[2][GEMM_TM][34];
    __shared__ __bf16 sB[2][32][34];

    const int tid  = threadIdx.x;
    const int lane = tid & 31;
    const int wave = tid >> 5;
    const int lm   = lane & 15;
    const int lh   = lane >> 4;

    const int rowBase = blockIdx.x * GEMM_TM;
    const int colBase = blockIdx.y * 32;
    const int waveRow = wave << 4;

    v8f acc0 = {}, acc1 = {};
    const int nk = (K + 31) >> 5;

    stage_tiles<HAS_SCALE>(A, rowScale, W, sA[0], sB[0], rowBase, colBase, 0, N, K, M, tid);

    for (int i = 0; i < nk; ++i) {
        __syncthreads();
        const int cur = i & 1;
        if (i + 1 < nk)
            stage_tiles<HAS_SCALE>(A, rowScale, W, sA[cur ^ 1], sB[cur ^ 1],
                                   rowBase, colBase, (i + 1) << 5, N, K, M, tid);

        union { v16bf v; unsigned u[8]; } fa, fb0, fb1;
#pragma unroll
        for (int v = 0; v < 8; ++v) {
            int k0 = ((v & 3) << 1) + ((v >> 2) << 4) + (lh << 3);
            fa.u[v] = *(const unsigned*)&sA[cur][waveRow + lm][k0];
        }
#pragma unroll
        for (int v = 0; v < 8; ++v) {
            int k0 = ((v & 3) << 1) + ((v >> 2) << 4) + (lh << 3);
            fb0.v[2 * v]     = sB[cur][k0][lm];
            fb0.v[2 * v + 1] = sB[cur][k0 + 1][lm];
            fb1.v[2 * v]     = sB[cur][k0][16 + lm];
            fb1.v[2 * v + 1] = sB[cur][k0 + 1][16 + lm];
        }
        acc0 = __builtin_amdgcn_wmma_f32_16x16x32_bf16(
                   false, fa.v, false, fb0.v, (short)0, acc0, false, false);
        acc1 = __builtin_amdgcn_wmma_f32_16x16x32_bf16(
                   false, fa.v, false, fb1.v, (short)0, acc1, false, false);
    }

    // --- epilogue: bias + ReLU + strided store (concat half) ---------------
#pragma unroll
    for (int v = 0; v < 8; ++v) {
        int r = rowBase + waveRow + (lh << 3) + v;
        if (r < N) {
            int c0 = colBase + lm;
            if (c0 < M) OUT[(size_t)r * ldc + c0] = fmaxf(acc0[v] + bias[c0], 0.0f);
            int c1 = colBase + 16 + lm;
            if (c1 < M) OUT[(size_t)r * ldc + c1] = fmaxf(acc1[v] + bias[c1], 0.0f);
        }
    }
}

// ---------------------------------------------------------------------------
// Per-graph pooling: one block per graph; binary-search segment bounds in the
// sorted node_graph_index, then stream mean/min/max/sum over the segment.
// pool layout per graph row: [t*800 + {mean(0), min(d), max(2d), sum(3d)}]
// ---------------------------------------------------------------------------
__global__ void pool_kernel(const float* __restrict__ h,
                            const int* __restrict__ gidx,
                            int N, int d,
                            float* __restrict__ pool, int poolStride, int outOfs) {
    int g = blockIdx.x;
    __shared__ int sb[2];
    if (threadIdx.x < 2) {
        int target = g + (int)threadIdx.x;   // lower_bound(target)
        int lo = 0, hi = N;
        while (lo < hi) {
            int mid = (lo + hi) >> 1;
            if (gidx[mid] < target) lo = mid + 1; else hi = mid;
        }
        sb[threadIdx.x] = lo;
    }
    __syncthreads();
    int beg = sb[0], end = sb[1];
    int cnt = end - beg;
    float invc = 1.0f / fmaxf((float)cnt, 1.0f);

    for (int j = threadIdx.x; j < d; j += blockDim.x) {
        float s  = 0.0f;
        float mn =  INFINITY;
        float mx = -INFINITY;
        for (int i = beg; i < end; ++i) {
            float v = h[(size_t)i * d + j];
            s += v;
            mn = fminf(mn, v);
            mx = fmaxf(mx, v);
        }
        float* base = pool + (size_t)g * poolStride + outOfs;
        base[j]         = s * invc;  // mean
        base[d + j]     = mn;        // min
        base[2 * d + j] = mx;        // max
        base[3 * d + j] = s;         // sum
    }
}

// ---------------------------------------------------------------------------
// Tiny MLP head: pool[64,1600] @ w1 + b1 -> @ w2 + b2 -> @ w3 + b3 -> out[64,10]
// ---------------------------------------------------------------------------
__global__ void mlp_kernel(const float* __restrict__ pool,
                           const float* __restrict__ w1, const float* __restrict__ b1,
                           const float* __restrict__ w2, const float* __restrict__ b2,
                           const float* __restrict__ w3, const float* __restrict__ b3,
                           float* __restrict__ out) {
    __shared__ float h1[GNN_GRAPHS * 50];
    __shared__ float h2[GNN_GRAPHS * 10];

    for (int idx = threadIdx.x; idx < GNN_GRAPHS * 50; idx += blockDim.x) {
        int g = idx / 50, m = idx - g * 50;
        float s = b1[m];
        const float* pr = pool + (size_t)g * GNN_POOLDIM;
        for (int k = 0; k < GNN_POOLDIM; ++k) s += pr[k] * w1[k * 50 + m];
        h1[idx] = s;
    }
    __syncthreads();
    for (int idx = threadIdx.x; idx < GNN_GRAPHS * 10; idx += blockDim.x) {
        int g = idx / 10, m = idx - g * 10;
        float s = b2[m];
        for (int k = 0; k < 50; ++k) s += h1[g * 50 + k] * w2[k * 10 + m];
        h2[idx] = s;
    }
    __syncthreads();
    for (int idx = threadIdx.x; idx < GNN_GRAPHS * 10; idx += blockDim.x) {
        int g = idx / 10, m = idx - g * 10;
        float s = b3[m];
        for (int k = 0; k < 10; ++k) s += h2[g * 10 + k] * w3[k * 10 + m];
        out[idx] = s;
    }
}

// ---------------------------------------------------------------------------
// Host-side orchestration
// ---------------------------------------------------------------------------
extern "C" void kernel_launch(void* const* d_in, const int* in_sizes, int n_in,
                              void* d_out, int out_size, void* d_ws, size_t ws_size,
                              hipStream_t stream) {
    (void)in_sizes; (void)n_in; (void)out_size; (void)ws_size;

    const int N = GNN_N_NODES;
    const int E = GNN_N_EDGES;

    const float* x0   = (const float*)d_in[0];
    const float* x1   = (const float*)d_in[1];
    const int*   edge = (const int*)d_in[2];
    const int*   row  = edge;           // edge_index[0]
    const int*   col  = edge + E;       // edge_index[1]
    const int*   gidx = (const int*)d_in[3];

    // params0: d_in[4..15], params1: d_in[16..27] as (Ws, Wn, b) x 4 layers
    const float* Ws[2][4]; const float* Wn[2][4]; const float* Bb[2][4];
    for (int t = 0; t < 2; ++t)
        for (int l = 0; l < 4; ++l) {
            int base = 4 + t * 12 + l * 3;
            Ws[t][l] = (const float*)d_in[base];
            Wn[t][l] = (const float*)d_in[base + 1];
            Bb[t][l] = (const float*)d_in[base + 2];
        }
    const float* w1 = (const float*)d_in[28];
    const float* b1 = (const float*)d_in[29];
    const float* w2 = (const float*)d_in[30];
    const float* b2 = (const float*)d_in[31];
    const float* w3 = (const float*)d_in[32];
    const float* b3 = (const float*)d_in[33];

    // ---- workspace layout -------------------------------------------------
    char* ws = (char*)d_ws;
    size_t off = 0;
    auto take = [&](size_t bytes) -> float* {
        float* p = (float*)(ws + off);
        off += (bytes + 255) & ~(size_t)255;
        return p;
    };
    float* buf0 = take((size_t)N * 200 * sizeof(float));   // ping
    float* agg  = take((size_t)N * 200 * sizeof(float));   // aggregation sums
    float* buf2 = take((size_t)N * 200 * sizeof(float));   // pong
    float* deg  = take((size_t)N * sizeof(float));
    float* dinv = take((size_t)N * sizeof(float));
    float* pool = take((size_t)GNN_GRAPHS * GNN_POOLDIM * sizeof(float));

    // ---- degrees (once) ---------------------------------------------------
    {
        long long n = N;
        fill_f32_kernel<<<(int)((n + 255) / 256), 256, 0, stream>>>(deg, 0.0f, n);
        deg_count_kernel<<<(E + 255) / 256, 256, 0, stream>>>(col, deg, E);
        deg_inv_kernel<<<(N + 255) / 256, 256, 0, stream>>>(deg, dinv, N);
    }

    static const int UNITS[4] = {100, 150, 200, 200};

    for (int t = 0; t < 2; ++t) {
        const float* hcur = (t == 0) ? x0 : x1;
        int d = GNN_D_IN;
        float* nxt   = buf0;
        float* other = buf2;

        for (int l = 0; l < 4; ++l) {
            int u = UNITS[l];
            int half = u / 2;

            // zero agg and scatter-add neighbor features (float2 lanes)
            long long nagg = (long long)N * d;
            fill_f32_kernel<<<(int)((nagg + 255) / 256), 256, 0, stream>>>(agg, 0.0f, nagg);
            int d2 = d >> 1;
            int total2 = E * d2;
            aggregate_kernel<<<(total2 + 255) / 256, 256, 0, stream>>>(hcur, row, col, agg, d2, total2);

            dim3 grid((N + GEMM_TM - 1) / GEMM_TM, (half + 31) / 32);
            // self half: cols [0, half)
            sage_gemm_bias_relu_kernel<false><<<grid, 256, 0, stream>>>(
                hcur, nullptr, Ws[t][l], Bb[t][l], nxt, N, d, half, u);
            // neighbor-mean half: cols [half, u), deg^-1 fused into A staging
            sage_gemm_bias_relu_kernel<true><<<grid, 256, 0, stream>>>(
                agg, dinv, Wn[t][l], Bb[t][l] + half, nxt + half, N, d, half, u);

            hcur = nxt;
            float* tmp = nxt; nxt = other; other = tmp;
            d = u;
        }

        // pools for this tower -> pool[:, t*800 .. t*800+800)
        pool_kernel<<<GNN_GRAPHS, 256, 0, stream>>>(hcur, gidx, N, d, pool, GNN_POOLDIM, t * 800);
    }

    mlp_kernel<<<1, 256, 0, stream>>>(pool, w1, b1, w2, b2, w3, b3, (float*)d_out);
}
